// GraphNetWork_84284438217071
// MI455X (gfx1250) — matrix operand order repaired and checked by reading, not compiled, not run on hardware
//
#include <hip/hip_runtime.h>
#include <math.h>

// ---------------- problem constants (match reference) ----------------
#define Hdim   512
#define LAYERS 2
#define OUTD   128
#define NEH    40000
#define NET    40000
#define NREL   3000
#define NEDGE  60000

// ---------------- CDNA5 WMMA types ----------------
typedef __attribute__((ext_vector_type(16))) __bf16 bf16x16;
typedef __attribute__((ext_vector_type(8)))  float  v8f;

union Frag {
  uint4   u[2];
  bf16x16 v;
};

static __device__ __forceinline__ unsigned short f2bf(float f) {
  unsigned int u = __float_as_uint(f);
  u += 0x7FFFu + ((u >> 16) & 1u);          // round-to-nearest-even
  return (unsigned short)(u >> 16);
}
static __device__ __forceinline__ unsigned int pack2(float a, float b) {
  return (unsigned int)f2bf(a) | ((unsigned int)f2bf(b) << 16);
}

// =====================================================================
// Weight pre-pack: f32 [K][N] -> bf16 tile-packed fragment-ready layout
//   tile (kt = k/32, nt = n/128); within tile: [n_local(128)][k_local(32)]
// Done once per launch; packed weights (~10.5 MB) stay hot in L2.
// =====================================================================
__global__ __launch_bounds__(256) void pack_w_kernel(
    const float* __restrict__ W, unsigned short* __restrict__ PW, int K, int N)
{
  int idx = blockIdx.x * 256 + threadIdx.x;
  if (idx >= K * N) return;
  int k = idx / N, n = idx % N;
  int kt = k >> 5, kl = k & 31;
  int nt = n >> 7, nl = n & 127;
  size_t dst = (((size_t)kt * (N >> 7) + nt) * 128 + nl) * 32 + kl;
  PW[dst] = f2bf(W[idx]);
}

// =====================================================================
// Generic tiled bf16-WMMA GEMM:  Cout = act( A' @ W + bias + Cin )
//   A'(m,k) = A[(k>>ksegShift)*segStride + m*lda + (k & ((1<<ksegShift)-1))]
//             * (rowDiv ? 1/max(rowDiv[m],1) : 1)
//   ksegShift = 9 for the concat-snapshot heads (Kseg=512); 30 otherwise
//   (seg==0, mask passes k through) — avoids runtime integer division.
//   PW is the tile-packed bf16 weight (see pack_w_kernel).
// Block tile 128x128, K step 32, 8 waves (4M x 2N), wave tile 32x64.
// =====================================================================
__global__ __launch_bounds__(256) void gemm_bf16_kernel(
    const float* __restrict__ A, int lda, int ksegShift, long long segStride,
    const float* __restrict__ rowDiv,
    const unsigned short* __restrict__ PW,
    const float* __restrict__ bias,
    const float* __restrict__ Cin,
    float* __restrict__ Cout,
    int M, int N, int K, int relu)
{
  __shared__ unsigned short As[128][32];
  __shared__ unsigned short Bs[128][32];   // [n][k] fragment-ready

  const int tid  = threadIdx.x;
  const int bm   = blockIdx.y * 128;
  const int bn   = blockIdx.x * 128;
  const int lane = tid & 31;
  const int wave = tid >> 5;
  const int wm   = (wave & 3) * 32;
  const int wn   = (wave >> 2) * 64;
  const int lm   = lane & 15;
  const int lhi  = lane >> 4;

  const int arow  = tid >> 1;
  const int ahalf = (tid & 1) * 16;
  const int ntiles  = N >> 7;
  const int ksegMask = (1 << ksegShift) - 1;

  // ---- loop-invariant A-staging state (hoisted out of the K loop)
  const int  am      = bm + arow;
  const bool mvalid  = (am < M);
  float      ascale  = 1.0f;
  if (mvalid && rowDiv) ascale = 1.0f / fmaxf(rowDiv[am], 1.0f);
  const float* arowp = A + (size_t)(mvalid ? am : 0) * lda;

  v8f acc[2][4] = {};

  for (int k0 = 0; k0 < K; k0 += 32) {
    // ---- stage A tile (f32 -> bf16 packed, 2x ds_store_b128 per thread)
    {
      int gk0 = k0 + ahalf;                 // 16-chunk never crosses a Kseg boundary
      uint4 p0, p1;
      if (mvalid) {
        int seg  = gk0 >> ksegShift;        // shift+and, no integer division
        int koff = gk0 & ksegMask;
        const float* src = arowp + (long long)seg * segStride + koff;
        const float s = ascale;
        float4 f0 = ((const float4*)src)[0];
        float4 f1 = ((const float4*)src)[1];
        float4 f2 = ((const float4*)src)[2];
        float4 f3 = ((const float4*)src)[3];
        if (k0 + 32 < K) __builtin_prefetch(src + 32, 0, 0);   // global_prefetch_b8
        p0.x = pack2(f0.x*s, f0.y*s); p0.y = pack2(f0.z*s, f0.w*s);
        p0.z = pack2(f1.x*s, f1.y*s); p0.w = pack2(f1.z*s, f1.w*s);
        p1.x = pack2(f2.x*s, f2.y*s); p1.y = pack2(f2.z*s, f2.w*s);
        p1.z = pack2(f3.x*s, f3.y*s); p1.w = pack2(f3.z*s, f3.w*s);
      } else {
        p0 = make_uint4(0, 0, 0, 0);
        p1 = make_uint4(0, 0, 0, 0);
      }
      *(uint4*)&As[arow][ahalf]     = p0;
      *(uint4*)&As[arow][ahalf + 8] = p1;
    }
    // ---- stage B tile: contiguous copy of packed weights (already [n][k])
    {
      size_t tbase = ((size_t)(k0 >> 5) * ntiles + (bn >> 7)) * (128 * 32);
      const uint4* src = (const uint4*)(PW + tbase) + (size_t)tid * 2;
      uint4* dst = (uint4*)(&Bs[0][0]) + (size_t)tid * 2;
      uint4 u0 = src[0];
      uint4 u1 = src[1];
      dst[0] = u0;
      dst[1] = u1;
    }
    __syncthreads();

    // ---- WMMA compute: fragments via b128 LDS loads (ISA 7.12.2 layout)
    Frag a0, a1;
    {
      int r0 = wm + lm, r1 = wm + 16 + lm;
      a0.u[0] = *(const uint4*)&As[r0][lhi*8];
      a0.u[1] = *(const uint4*)&As[r0][lhi*8 + 16];
      a1.u[0] = *(const uint4*)&As[r1][lhi*8];
      a1.u[1] = *(const uint4*)&As[r1][lhi*8 + 16];
    }
    #pragma unroll
    for (int nt = 0; nt < 4; ++nt) {
      Frag b;
      int n = wn + nt*16 + lm;
      b.u[0] = *(const uint4*)&Bs[n][lhi*8];
      b.u[1] = *(const uint4*)&Bs[n][lhi*8 + 16];
      acc[0][nt] = __builtin_amdgcn_wmma_f32_16x16x32_bf16(
          false, a0.v, false, b.v, (short)0, acc[0][nt], false, false);
      acc[1][nt] = __builtin_amdgcn_wmma_f32_16x16x32_bf16(
          false, a1.v, false, b.v, (short)0, acc[1][nt], false, false);
    }
    __syncthreads();
  }

  // ---- epilogue: bias + residual + relu
  #pragma unroll
  for (int mt = 0; mt < 2; ++mt) {
    #pragma unroll
    for (int r = 0; r < 8; ++r) {
      int m = bm + wm + mt*16 + lhi*8 + r;
      if (m >= M) continue;
      #pragma unroll
      for (int nt = 0; nt < 4; ++nt) {
        int n = bn + wn + nt*16 + lm;
        float v = acc[mt][nt][r];
        if (bias) v += bias[n];
        if (Cin)  v += Cin[(size_t)m * N + n];
        if (relu) v = fmaxf(v, 0.0f);
        Cout[(size_t)m * N + n] = v;
      }
    }
  }
}

// =====================================================================
// Edge MLP: d[e] = sigmoid( relu([hs|hd] @ W1 + b1) @ W2 + b2 )
// Gathered A (K=1024: 512 src + 512 dst), N=128, then per-edge reduction.
// W1 pre-packed. Also accumulates sum(d^2) into dsq.
// =====================================================================
__global__ __launch_bounds__(256) void edge_mlp_kernel(
    const float* __restrict__ featS, const float* __restrict__ featD,
    const int* __restrict__ si, const int* __restrict__ di,
    const unsigned short* __restrict__ PW1, const float* __restrict__ b1,
    const float* __restrict__ W2, const float* __restrict__ b2,
    float* __restrict__ d_e, float* __restrict__ dsq, int E)
{
  __shared__ unsigned short As[128][32];
  __shared__ unsigned short Bs[128][32];
  __shared__ float hid[128][128];
  __shared__ float red[128];

  const int tid  = threadIdx.x;
  const int be   = blockIdx.x * 128;
  const int lane = tid & 31;
  const int wave = tid >> 5;
  const int wm   = (wave & 3) * 32;
  const int wn   = (wave >> 2) * 64;
  const int lm   = lane & 15;
  const int lhi  = lane >> 4;

  const int arow  = tid >> 1;
  const int ahalf = (tid & 1) * 16;

  // ---- loop-invariant gather state (index loads hoisted out of K loop)
  const int  ae     = be + arow;
  const bool evalid = (ae < E);
  const float* rowS = featS;
  const float* rowD = featD;
  if (evalid) {
    rowS = featS + (size_t)si[ae] * Hdim;
    rowD = featD + (size_t)di[ae] * Hdim;
  }

  v8f acc[2][4] = {};

  for (int k0 = 0; k0 < 1024; k0 += 32) {
    { // gathered A staging (packed bf16, 2x ds_store_b128)
      int gk0 = k0 + ahalf;
      uint4 p0, p1;
      if (evalid) {
        const float* src = (gk0 < 512) ? (rowS + gk0) : (rowD + (gk0 - 512));
        float4 f0 = ((const float4*)src)[0];
        float4 f1 = ((const float4*)src)[1];
        float4 f2 = ((const float4*)src)[2];
        float4 f3 = ((const float4*)src)[3];
        p0.x = pack2(f0.x, f0.y); p0.y = pack2(f0.z, f0.w);
        p0.z = pack2(f1.x, f1.y); p0.w = pack2(f1.z, f1.w);
        p1.x = pack2(f2.x, f2.y); p1.y = pack2(f2.z, f2.w);
        p1.z = pack2(f3.x, f3.y); p1.w = pack2(f3.z, f3.w);
      } else {
        p0 = make_uint4(0, 0, 0, 0);
        p1 = make_uint4(0, 0, 0, 0);
      }
      *(uint4*)&As[arow][ahalf]     = p0;
      *(uint4*)&As[arow][ahalf + 8] = p1;
    }
    { // W1 staging: contiguous packed copy (N=128 -> single n-tile)
      size_t tbase = (size_t)(k0 >> 5) * (128 * 32);
      const uint4* src = (const uint4*)(PW1 + tbase) + (size_t)tid * 2;
      uint4* dst = (uint4*)(&Bs[0][0]) + (size_t)tid * 2;
      uint4 u0 = src[0];
      uint4 u1 = src[1];
      dst[0] = u0;
      dst[1] = u1;
    }
    __syncthreads();
    Frag a0, a1;
    {
      int r0 = wm + lm, r1 = wm + 16 + lm;
      a0.u[0] = *(const uint4*)&As[r0][lhi*8];
      a0.u[1] = *(const uint4*)&As[r0][lhi*8 + 16];
      a1.u[0] = *(const uint4*)&As[r1][lhi*8];
      a1.u[1] = *(const uint4*)&As[r1][lhi*8 + 16];
    }
    #pragma unroll
    for (int nt = 0; nt < 4; ++nt) {
      Frag b;
      int n = wn + nt*16 + lm;
      b.u[0] = *(const uint4*)&Bs[n][lhi*8];
      b.u[1] = *(const uint4*)&Bs[n][lhi*8 + 16];
      acc[0][nt] = __builtin_amdgcn_wmma_f32_16x16x32_bf16(
          false, a0.v, false, b.v, (short)0, acc[0][nt], false, false);
      acc[1][nt] = __builtin_amdgcn_wmma_f32_16x16x32_bf16(
          false, a1.v, false, b.v, (short)0, acc[1][nt], false, false);
    }
    __syncthreads();
  }

  // hid = relu(acc + b1) -> LDS
  #pragma unroll
  for (int mt = 0; mt < 2; ++mt)
    #pragma unroll
    for (int r = 0; r < 8; ++r) {
      int m = wm + mt*16 + lhi*8 + r;
      #pragma unroll
      for (int nt = 0; nt < 4; ++nt) {
        int n = wn + nt*16 + lm;
        hid[m][n] = fmaxf(acc[mt][nt][r] + b1[n], 0.0f);
      }
    }
  __syncthreads();

  // per-edge: sigmoid(hid @ W2 + b2); accumulate d^2
  if (tid < 128) {
    float d2 = 0.0f;
    int e = be + tid;
    if (e < E) {
      float sum = b2[0];
      #pragma unroll 8
      for (int j = 0; j < 128; ++j) sum += hid[tid][j] * W2[j];
      float d = 1.0f / (1.0f + __expf(-sum));
      d_e[e] = d;
      d2 = d * d;
    }
    red[tid] = d2;
  }
  __syncthreads();
  for (int o = 64; o > 0; o >>= 1) {
    if (tid < o) red[tid] += red[tid + o];
    __syncthreads();
  }
  if (tid == 0) atomicAdd(dsq, red[0]);
}

// =====================================================================
// Scatter: msg[dst] += hs * (edge_w + d); deg[dst] += 1   (one block/edge)
// =====================================================================
__global__ __launch_bounds__(128) void scatter_kernel(
    const float* __restrict__ featS,
    const int* __restrict__ si, const int* __restrict__ di,
    const float* __restrict__ ew, const float* __restrict__ d_e,
    float* __restrict__ msg, float* __restrict__ deg, int E)
{
  int e = blockIdx.x;
  if (e >= E) return;
  int t = threadIdx.x;               // 128 threads * float4 = 512 floats
  float w = ew[e] + d_e[e];
  const float4* src = (const float4*)(featS + (size_t)si[e] * Hdim);
  float* dst = msg + (size_t)di[e] * Hdim;
  float4 v = src[t];
  atomicAdd(dst + t*4 + 0, v.x * w);
  atomicAdd(dst + t*4 + 1, v.y * w);
  atomicAdd(dst + t*4 + 2, v.z * w);
  atomicAdd(dst + t*4 + 3, v.w * w);
  if (t == 0) atomicAdd(deg + di[e], 1.0f);
}

// =====================================================================
// LayerNorm over `width`, optional split of output at `half`.
// =====================================================================
__global__ __launch_bounds__(256) void ln_kernel(
    const float* __restrict__ X, int width,
    const float* __restrict__ g, const float* __restrict__ b,
    float* __restrict__ out0, float* __restrict__ out1, int half)
{
  __shared__ float red[256];
  int row = blockIdx.x;
  int t = threadIdx.x;
  const float* x = X + (size_t)row * width;

  float s = 0.0f;
  for (int c = t; c < width; c += 256) s += x[c];
  red[t] = s; __syncthreads();
  for (int o = 128; o > 0; o >>= 1) { if (t < o) red[t] += red[t+o]; __syncthreads(); }
  float mean = red[0] / (float)width;
  __syncthreads();

  float v = 0.0f;
  for (int c = t; c < width; c += 256) { float d = x[c] - mean; v += d*d; }
  red[t] = v; __syncthreads();
  for (int o = 128; o > 0; o >>= 1) { if (t < o) red[t] += red[t+o]; __syncthreads(); }
  float rstd = rsqrtf(red[0] / (float)width + 1e-5f);
  __syncthreads();

  for (int c = t; c < width; c += 256) {
    float y = (x[c] - mean) * rstd * g[c] + b[c];
    if (c < half) out0[(size_t)row * half + c] = y;
    else          out1[(size_t)row * half + (c - half)] = y;
  }
}

// =====================================================================
// Final ada_loss = (sum_s sqrt(dsq[s]/E)) / 4 / L * 0.05
// =====================================================================
__global__ void ada_kernel(const float* __restrict__ dsq, float* __restrict__ out)
{
  float a = 0.0f;
  for (int s = 0; s < 8; ++s) a += sqrtf(dsq[s] / (float)NEDGE);
  out[0] = a / 4.0f / (float)LAYERS * 0.05f;
}

// =====================================================================
// Host orchestration
// =====================================================================
static inline void launch_gemm(const float* A, int lda, int ksegShift, long long segStride,
                               const float* rowDiv, const unsigned short* PW,
                               const float* bias, const float* Cin, float* Cout,
                               int M, int N, int K, int relu, hipStream_t s)
{
  dim3 grid(N / 128, (M + 127) / 128);
  gemm_bf16_kernel<<<grid, 256, 0, s>>>(A, lda, ksegShift, segStride, rowDiv,
                                        PW, bias, Cin, Cout, M, N, K, relu);
}

extern "C" void kernel_launch(void* const* d_in, const int* in_sizes, int n_in,
                              void* d_out, int out_size, void* d_ws, size_t ws_size,
                              hipStream_t stream) {
  (void)in_sizes; (void)n_in; (void)out_size; (void)ws_size;

  const float* rel_embeds  = (const float*)d_in[0];
  const float* ent_embeds  = (const float*)d_in[1];
  const int*   src_idx     = (const int*)  d_in[2];
  const int*   dst_idx     = (const int*)  d_in[3];
  const float* edge_w      = (const float*)d_in[4];
  const float* rel_proj_W  = (const float*)d_in[5];
  const float* rel_proj_b  = (const float*)d_in[6];
  const float* rel_ln_g    = (const float*)d_in[7];
  const float* rel_ln_b    = (const float*)d_in[8];
  const float* ent_proj_W  = (const float*)d_in[9];
  const float* ent_proj_b  = (const float*)d_in[10];
  const float* ent_ln_g    = (const float*)d_in[11];
  const float* ent_ln_b    = (const float*)d_in[12];
  const float* conv_weight = (const float*)d_in[13];
  const float* conv_bias   = (const float*)d_in[14];
  const float* loop_weight = (const float*)d_in[15];
  const float* wp_W1       = (const float*)d_in[16];
  const float* wp_b1       = (const float*)d_in[17];
  const float* wp_W2       = (const float*)d_in[18];
  const float* wp_b2       = (const float*)d_in[19];
  const float* hout_W      = (const float*)d_in[20];
  const float* hout_b      = (const float*)d_in[21];
  const float* tout_W      = (const float*)d_in[22];
  const float* tout_b      = (const float*)d_in[23];
  float* out = (float*)d_out;

  // -------- workspace layout (floats) --------
  float* ws = (float*)d_ws;
  size_t off = 0;
  float* eh_snap  = ws + off; off += (size_t)3 * NEH  * Hdim;  // [3][NEH][H]
  float* et_snap  = ws + off; off += (size_t)3 * NET  * Hdim;
  float* rel_snap = ws + off; off += (size_t)3 * NREL * Hdim;
  float* agg_eh   = ws + off; off += (size_t)NEH  * Hdim;
  float* agg_et   = ws + off; off += (size_t)NET  * Hdim;
  float* agg_rel  = ws + off; off += (size_t)NREL * Hdim;
  float* msg      = ws + off; off += (size_t)NEH  * Hdim;      // max n_dst
  float* deg      = ws + off; off += NEH;
  float* d_e      = ws + off; off += NEDGE;
  float* dsq      = ws + off; off += 8;

  // packed bf16 weights (ushort), after the float region
  unsigned short* pwbase = (unsigned short*)(ws + off);
  size_t poff = 0;
  unsigned short* pw_rel  = pwbase + poff; poff += (size_t)768 * 512;
  unsigned short* pw_ent  = pwbase + poff; poff += (size_t)768 * 1024;
  unsigned short* pw_conv = pwbase + poff; poff += (size_t)8 * Hdim * Hdim;
  unsigned short* pw_loop = pwbase + poff; poff += (size_t)2 * Hdim * Hdim;
  unsigned short* pw_W1   = pwbase + poff; poff += (size_t)8 * 1024 * 128;
  unsigned short* pw_hout = pwbase + poff; poff += (size_t)3 * Hdim * OUTD;
  unsigned short* pw_tout = pwbase + poff; poff += (size_t)3 * Hdim * OUTD;

  // pre-LN projection buffers alias agg (disjoint lifetimes):
  float* tmp_ent = agg_eh;    // [NEH][1024] contiguous over agg_eh|agg_et
  float* tmp_rel = agg_rel;   // [NREL][512]

  hipMemsetAsync(dsq, 0, 8 * sizeof(float), stream);

  // -------- pack all weights to bf16 tile layout (once, tiny) --------
  {
    auto pack = [&](const float* W, unsigned short* PW, int K, int N) {
      int total = K * N;
      pack_w_kernel<<<(total + 255) / 256, 256, 0, stream>>>(W, PW, K, N);
    };
    pack(rel_proj_W, pw_rel, 768, 512);
    pack(ent_proj_W, pw_ent, 768, 1024);
    for (int j = 0; j < 8; ++j)
      pack(conv_weight + (size_t)j * Hdim * Hdim, pw_conv + (size_t)j * Hdim * Hdim, Hdim, Hdim);
    for (int l = 0; l < 2; ++l)
      pack(loop_weight + (size_t)l * Hdim * Hdim, pw_loop + (size_t)l * Hdim * Hdim, Hdim, Hdim);
    for (int j = 0; j < 8; ++j)
      pack(wp_W1 + (size_t)j * 1024 * 128, pw_W1 + (size_t)j * 1024 * 128, 1024, 128);
    pack(hout_W, pw_hout, 3 * Hdim, OUTD);
    pack(tout_W, pw_tout, 3 * Hdim, OUTD);
  }

  // -------- input projections + LN --------
  launch_gemm(rel_embeds, 768, 30, 0, nullptr, pw_rel, rel_proj_b,
              nullptr, tmp_rel, NREL, 512, 768, 1, stream);
  launch_gemm(ent_embeds, 768, 30, 0, nullptr, pw_ent, ent_proj_b,
              nullptr, tmp_ent, NEH, 1024, 768, 1, stream);
  ln_kernel<<<NREL, 256, 0, stream>>>(tmp_rel, 512, rel_ln_g, rel_ln_b,
                                      rel_snap, rel_snap, 512);
  ln_kernel<<<NEH, 256, 0, stream>>>(tmp_ent, 1024, ent_ln_g, ent_ln_b,
                                     eh_snap, et_snap, 512);

  const int SRC_T[4] = {2, 2, 0, 1};
  const int DST_T[4] = {0, 1, 2, 2};
  const int NN[3]    = {NEH, NET, NREL};
  float* snaps[3] = {eh_snap, et_snap, rel_snap};
  float* aggs[3]  = {agg_eh, agg_et, agg_rel};

  // -------- GCN layers --------
  for (int l = 0; l < LAYERS; ++l) {
    hipMemsetAsync(agg_eh, 0, (size_t)(NEH + NET + NREL) * Hdim * sizeof(float), stream);
    for (int i = 0; i < 4; ++i) {
      const float* fS = snaps[SRC_T[i]] + (size_t)l * NN[SRC_T[i]] * Hdim;
      const float* fD = snaps[DST_T[i]] + (size_t)l * NN[DST_T[i]] * Hdim;
      const int* si = src_idx + (size_t)i * NEDGE;
      const int* di = dst_idx + (size_t)i * NEDGE;
      const int il = i * LAYERS + l;

      edge_mlp_kernel<<<(NEDGE + 127) / 128, 256, 0, stream>>>(
          fS, fD, si, di,
          pw_W1 + (size_t)il * 1024 * 128, wp_b1 + (size_t)il * 128,
          wp_W2 + (size_t)il * 128,        wp_b2 + il,
          d_e, dsq + (l * 4 + i), NEDGE);

      const int ndst = NN[DST_T[i]];
      hipMemsetAsync(msg, 0, (size_t)ndst * Hdim * sizeof(float), stream);
      hipMemsetAsync(deg, 0, (size_t)ndst * sizeof(float), stream);
      scatter_kernel<<<NEDGE, 128, 0, stream>>>(fS, si, di,
          edge_w + (size_t)i * NEDGE, d_e, msg, deg, NEDGE);

      // agg[DST] += (msg / clip(deg,1)) @ conv_weight[l,i]
      launch_gemm(msg, Hdim, 30, 0, deg,
                  pw_conv + (size_t)(l * 4 + i) * Hdim * Hdim, nullptr,
                  aggs[DST_T[i]], aggs[DST_T[i]], ndst, Hdim, Hdim, 0, stream);
    }
    // feats_new[t] = relu(agg[t] + feats[t] @ loop_W[l] + conv_bias[l])
    for (int t = 0; t < 3; ++t) {
      launch_gemm(snaps[t] + (size_t)l * NN[t] * Hdim, Hdim, 30, 0, nullptr,
                  pw_loop + (size_t)l * Hdim * Hdim,
                  conv_bias + (size_t)l * Hdim,
                  aggs[t], snaps[t] + (size_t)(l + 1) * NN[t] * Hdim,
                  NN[t], Hdim, Hdim, 1, stream);
    }
  }

  // -------- output heads over concatenated snapshots (Kseg=512 -> shift 9) --------
  launch_gemm(eh_snap, Hdim, 9, (long long)NEH * Hdim, nullptr,
              pw_hout, hout_b, nullptr, out, NEH, OUTD, 3 * Hdim, 0, stream);
  launch_gemm(et_snap, Hdim, 9, (long long)NET * Hdim, nullptr,
              pw_tout, tout_b, nullptr, out + (size_t)NEH * OUTD,
              NET, OUTD, 3 * Hdim, 0, stream);

  ada_kernel<<<1, 1, 0, stream>>>(dsq, out + (size_t)2 * NEH * OUTD);
}
